// LiquidityResidualBackbone_17763984736345
// MI455X (gfx1250) — matrix operand order repaired and checked by reading, not compiled, not run on hardware
//
#include <hip/hip_runtime.h>
#include <hip/hip_bf16.h>
#include <math.h>

#define NN 100000
#define DD 256
#define HH 8
#define BB 4096
#define TT 262144
#define DHH 32
#define HIDD 256
#define NQQ 3

typedef __attribute__((ext_vector_type(16))) __bf16 v16bf;
typedef __attribute__((ext_vector_type(8)))  float  v8f;

union BF16x16 { v16bf v; unsigned short u[16]; };

__device__ __forceinline__ unsigned short f2bf(float f) {
    unsigned int u = __float_as_uint(f);
    unsigned int r = u + 0x7fffu + ((u >> 16) & 1u);   // round-to-nearest-even
    return (unsigned short)(r >> 16);
}

// order-preserving float<->uint keys for atomicMax on signed floats
__device__ __forceinline__ unsigned int fkey(float f) {
    unsigned int u = __float_as_uint(f);
    return (u & 0x80000000u) ? ~u : (u | 0x80000000u);
}
__device__ __forceinline__ float funkey(unsigned int k) {
    unsigned int u = (k & 0x80000000u) ? (k & 0x7fffffffu) : ~k;
    return __uint_as_float(u);
}

// ---------------- K0: init scratch accumulators ----------------
__global__ void k_init(unsigned int* segmax, float* den, float* ctx) {
    int gid = blockIdx.x * blockDim.x + threadIdx.x;      // grid = B*2*D threads
    if (gid < BB * 16) { segmax[gid] = 0u; den[gid] = 0.0f; }
    ctx[gid] = 0.0f;                                      // B*2*256 elements
}

// ---------------- K0b: pack Wv matrices into WMMA bf16 B-operand layout ----
// B tile (32x16): lanes 0-15 -> N=lane, K=kt*32+0..15 ; lanes 16-31 -> N=lane-16, K=kt*32+16..31
__global__ void k_pack(const float* __restrict__ pma_Wv,
                       const float* __restrict__ cr_Wv,
                       unsigned short* __restrict__ packed) {
    int gid = blockIdx.x * blockDim.x + threadIdx.x;      // 2*8*16*32*16 = 131072
    int e    = gid & 15;
    int lane = (gid >> 4) & 31;
    int nt   = (gid >> 9) & 15;
    int kt   = (gid >> 13) & 7;
    int mat  = (gid >> 16) & 1;
    int n = nt * 16 + (lane & 15);
    int k = kt * 32 + (lane >> 4) * 16 + e;
    const float* W = mat ? cr_Wv : pma_Wv;
    packed[gid] = f2bf(W[k * DD + n]);
}

// ---------------- K1: gather targets + q2 = targets @ cr_Wq ----------------
__global__ void k_targets_q2(const float* __restrict__ emb, const int* __restrict__ tidx,
                             const float* __restrict__ crWq,
                             float* __restrict__ targets, float* __restrict__ q2) {
    __shared__ float sx[DD];
    int b = blockIdx.x, tid = threadIdx.x;
    int idx = tidx[b];
    float x = emb[(size_t)idx * DD + tid];
    sx[tid] = x;
    targets[b * DD + tid] = x;
    __syncthreads();
    float acc = 0.0f;
    for (int c = 0; c < DD; c++) acc += sx[c] * crWq[c * DD + tid];
    q2[b * DD + tid] = acc;
}

// ---------------- K2: fold PMA query into Wk -> Weff_pma [8][256] ----------
__global__ void k_weffp(const float* __restrict__ seed, const float* __restrict__ Wq,
                        const float* __restrict__ Wk, float* __restrict__ weffp) {
    __shared__ float ss[DD];
    __shared__ float sq[DD];
    int tid = threadIdx.x;
    ss[tid] = seed[tid];
    __syncthreads();
    float acc = 0.0f;
    for (int c = 0; c < DD; c++) acc += ss[c] * Wq[c * DD + tid];
    sq[tid] = acc;           // q = seed @ pma_Wq
    __syncthreads();
    for (int h = 0; h < HH; h++) {
        float a = 0.0f;
        for (int d = 0; d < DHH; d++) a += Wk[tid * DD + h * DHH + d] * sq[h * DHH + d];
        weffp[h * DD + tid] = a;
    }
}

// ---------------- K3: per-batch folded cross query: Weff2 [B][8][256] ------
__global__ void k_weff2(const float* __restrict__ q2, const float* __restrict__ crWk,
                        float* __restrict__ weff2) {
    __shared__ float sq[DD];
    int b = blockIdx.x, tid = threadIdx.x;
    sq[tid] = q2[b * DD + tid];
    __syncthreads();
    for (int h = 0; h < HH; h++) {
        float a = 0.0f;
        for (int d = 0; d < DHH; d++) a += crWk[tid * DD + h * DHH + d] * sq[h * DHH + d];
        weff2[((size_t)b * HH + h) * DD + tid] = a;
    }
}

// ---------------- K4: logits for both attentions + segment max -------------
__global__ void k_logits(const float* __restrict__ emb, const int* __restrict__ pidx,
                         const int* __restrict__ pbatch,
                         const float* __restrict__ weffp, const float* __restrict__ weff2,
                         float* __restrict__ ebuf, unsigned int* __restrict__ segmax) {
    int gid = blockIdx.x * blockDim.x + threadIdx.x;      // T*16
    int t = gid >> 4, j = gid & 15;
    int h = j & 7, mat = j >> 3;
    int seg = pbatch[t];
    const float* x = emb + (size_t)pidx[t] * DD;
    const float* w = mat ? (weff2 + ((size_t)seg * HH + h) * DD) : (weffp + h * DD);
    float a = 0.0f;
    for (int c = 0; c < DD; c++) a += x[c] * w[c];
    a *= 0.17677669529663687f;                            // 1/sqrt(32)
    ebuf[gid] = a;
    atomicMax(&segmax[seg * 16 + j], fkey(a));
}

// ---------------- K5: e = exp(l - m), segment denominators -----------------
__global__ void k_expsum(const int* __restrict__ pbatch, float* __restrict__ ebuf,
                         const unsigned int* __restrict__ segmax, float* __restrict__ den) {
    int gid = blockIdx.x * blockDim.x + threadIdx.x;      // T*16
    int t = gid >> 4, j = gid & 15;
    int seg = pbatch[t];
    float m = funkey(segmax[seg * 16 + j]);
    float e = __expf(ebuf[gid] - m);
    ebuf[gid] = e;
    atomicAdd(&den[seg * 16 + j], e);
}

// ---------------- K6: WMMA V-projections + weighted segment accumulation ---
// One wave per 16-token tile. A = X tile (16x256 bf16, 64 VGPRs), B from packed ws.
__global__ void __launch_bounds__(128)
k_wmma_v(const float* __restrict__ emb, const int* __restrict__ pidx,
         const int* __restrict__ pbatch, const float* __restrict__ pw,
         const unsigned short* __restrict__ packed, const float* __restrict__ ebuf,
         const float* __restrict__ den, float* __restrict__ ctx) {
    __shared__ int   sseg[4][16];
    __shared__ float scoef[4][16][16];
    int lane = threadIdx.x & 31;
    int wv   = threadIdx.x >> 5;
    int tile = blockIdx.x * 4 + wv;
    int t0   = tile * 16;

    // stage per-row softmax coefficients: coeff = port_weight * e / den
    if (lane < 16) {
        int t = t0 + lane;
        int s = pbatch[t];
        float w = pw[t];
        sseg[wv][lane] = s;
        for (int j = 0; j < 16; j++) {
            float d = den[s * 16 + j];
            d = d > 0.0f ? d : 1.0f;
            scoef[wv][lane][j] = w * ebuf[(size_t)t * 16 + j] / d;
        }
    }
    __syncthreads();

    // A operand: lane<16 -> row m=lane, K [c0,c0+16); lane>=16 -> +16
    int m = lane & 15, half = lane >> 4;
    const float* rowp = emb + (size_t)pidx[t0 + m] * DD;
    v16bf A[8];
    #pragma unroll
    for (int kt = 0; kt < 8; kt++) {
        int c0 = kt * 32 + half * 16;
        BF16x16 a;
        #pragma unroll
        for (int jj = 0; jj < 8; jj++) {
            a.u[jj]     = f2bf(rowp[c0 + jj]);
            a.u[8 + jj] = f2bf(rowp[c0 + 8 + jj]);
        }
        A[kt] = a.v;
    }

    // hoist per-row ctx base pointers (this lane's column m within each row's segment)
    float* ctxbase[8];
    #pragma unroll
    for (int r = 0; r < 8; r++)
        ctxbase[r] = ctx + (size_t)sseg[wv][half * 8 + r] * (2 * DD) + m;

    for (int mat = 0; mat < 2; mat++) {
        for (int nt = 0; nt < 16; nt++) {
            v8f acc = {};
            #pragma unroll
            for (int kt = 0; kt < 8; kt++) {
                v16bf bm = *((const v16bf*)(packed + (size_t)((mat * 8 + kt) * 16 + nt) * 512) + lane);
                acc = __builtin_amdgcn_wmma_f32_16x16x32_bf16(
                          false, A[kt], false, bm, (short)0, acc, false, false);
            }
            int head = nt >> 1;
            // batch the 8 coefficient LDS reads so they can clause
            float cf[8];
            #pragma unroll
            for (int r = 0; r < 8; r++)
                cf[r] = scoef[wv][half * 8 + r][mat * 8 + head];
            #pragma unroll
            for (int r = 0; r < 8; r++)
                atomicAdd(ctxbase[r] + mat * DD + nt * 16, acc[r] * cf[r]);
        }
    }
}

// ---------------- K7: Wo matvecs, residual, LayerNorm, MLPs, heads ---------
// Row-blocked by 8: one weight stream feeds 8 batch rows (8x less L2 traffic).
#define RB 8
__global__ void __launch_bounds__(256)
k_final(const float* __restrict__ targets, const float* __restrict__ ctx,
        const float* __restrict__ pma_Wo, const float* __restrict__ cr_Wo,
        const float* __restrict__ ln_g, const float* __restrict__ ln_b,
        const float* __restrict__ fW1, const float* __restrict__ fb1,
        const float* __restrict__ fW2, const float* __restrict__ fb2,
        const float* __restrict__ hW1, const float* __restrict__ hb1,
        const float* __restrict__ hW2, const float* __restrict__ hb2,
        float* __restrict__ out) {
    __shared__ float s_x[RB][DD], s_c1[RB][DD], s_c2[RB][DD];
    __shared__ float s_z[RB][3 * DD];
    __shared__ float s_h[RB][DD];
    __shared__ float s_v[RB][DD];
    int b0 = blockIdx.x * RB, tid = threadIdx.x;

    #pragma unroll
    for (int r = 0; r < RB; r++) {
        s_x[r][tid]  = targets[(b0 + r) * DD + tid];
        s_c1[r][tid] = ctx[((size_t)(b0 + r) * 2 + 0) * DD + tid];
        s_c2[r][tid] = ctx[((size_t)(b0 + r) * 2 + 1) * DD + tid];
    }
    __syncthreads();

    // contexts = c1 @ pma_Wo ; att = c2 @ cr_Wo  (one weight load -> 8 FMAs)
    float a1[RB] = {}, a2[RB] = {};
    for (int c = 0; c < DD; c++) {
        float w1 = pma_Wo[c * DD + tid], w2 = cr_Wo[c * DD + tid];
        #pragma unroll
        for (int r = 0; r < RB; r++) {
            a1[r] += s_c1[r][c] * w1;
            a2[r] += s_c2[r][c] * w2;
        }
    }
    #pragma unroll
    for (int r = 0; r < RB; r++) {
        s_z[r][tid]          = s_x[r][tid];
        s_z[r][DD + tid]     = a1[r];                       // contexts (0 for empty segments)
        s_z[r][2 * DD + tid] = s_x[r][tid] + a2[r];         // fused residual
    }
    __syncthreads();

    // LayerNorm(768) : one wave per row, wave32 shuffle reduction
    {
        int w = tid >> 5, lane = tid & 31;
        float s = 0.0f, s2 = 0.0f;
        for (int i = lane; i < 3 * DD; i += 32) {
            float v = s_z[w][i];
            s += v; s2 += v * v;
        }
        #pragma unroll
        for (int o = 16; o > 0; o >>= 1) {
            s  += __shfl_down(s,  o, 32);
            s2 += __shfl_down(s2, o, 32);
        }
        s  = __shfl(s,  0, 32);
        s2 = __shfl(s2, 0, 32);
        float mu  = s  * (1.0f / (3 * DD));
        float var = s2 * (1.0f / (3 * DD)) - mu * mu;
        float inv = rsqrtf(var + 1e-5f);
        for (int i = lane; i < 3 * DD; i += 32)
            s_z[w][i] = (s_z[w][i] - mu) * inv * ln_g[i] + ln_b[i];
    }
    __syncthreads();

    float acc[RB];
    // h1 = relu(z @ fuse_W1 + b1)
    #pragma unroll
    for (int r = 0; r < RB; r++) acc[r] = fb1[tid];
    for (int i = 0; i < 3 * DD; i++) {
        float wv_ = fW1[i * DD + tid];
        #pragma unroll
        for (int r = 0; r < RB; r++) acc[r] += s_z[r][i] * wv_;
    }
    #pragma unroll
    for (int r = 0; r < RB; r++) s_h[r][tid] = acc[r] > 0.0f ? acc[r] : 0.0f;
    __syncthreads();

    // z2 = h1 @ fuse_W2 + b2
    #pragma unroll
    for (int r = 0; r < RB; r++) acc[r] = fb2[tid];
    for (int c = 0; c < DD; c++) {
        float wv_ = fW2[c * DD + tid];
        #pragma unroll
        for (int r = 0; r < RB; r++) acc[r] += s_h[r][c] * wv_;
    }
    #pragma unroll
    for (int r = 0; r < RB; r++) s_v[r][tid] = acc[r];
    __syncthreads();

    // h2 = relu(z2 @ head_W1 + b1h)
    #pragma unroll
    for (int r = 0; r < RB; r++) acc[r] = hb1[tid];
    for (int c = 0; c < DD; c++) {
        float wv_ = hW1[c * HIDD + tid];
        #pragma unroll
        for (int r = 0; r < RB; r++) acc[r] += s_v[r][c] * wv_;
    }
    __syncthreads();                       // all reads of s_h above complete
    #pragma unroll
    for (int r = 0; r < RB; r++) s_h[r][tid] = acc[r] > 0.0f ? acc[r] : 0.0f;
    __syncthreads();

    // out = h2 @ head_W2 + b2h  (24 outputs per block)
    if (tid < RB * NQQ) {
        int r = tid / NQQ, q = tid % NQQ;
        float o = hb2[q];
        for (int c = 0; c < HIDD; c++) o += s_h[r][c] * hW2[c * NQQ + q];
        out[(b0 + r) * NQQ + q] = o;
    }
}

// ---------------- launch ----------------
extern "C" void kernel_launch(void* const* d_in, const int* in_sizes, int n_in,
                              void* d_out, int out_size, void* d_ws, size_t ws_size,
                              hipStream_t stream) {
    (void)in_sizes; (void)n_in; (void)out_size; (void)ws_size;
    const float* emb     = (const float*)d_in[0];
    const int*   tgt_idx = (const int*)d_in[1];
    const int*   p_idx   = (const int*)d_in[2];
    const int*   p_batch = (const int*)d_in[3];
    const float* p_w     = (const float*)d_in[4];
    const float* pma_seed= (const float*)d_in[5];
    const float* pma_Wq  = (const float*)d_in[6];
    const float* pma_Wk  = (const float*)d_in[7];
    const float* pma_Wv  = (const float*)d_in[8];
    const float* pma_Wo  = (const float*)d_in[9];
    const float* cr_Wq   = (const float*)d_in[10];
    const float* cr_Wk   = (const float*)d_in[11];
    const float* cr_Wv   = (const float*)d_in[12];
    const float* cr_Wo   = (const float*)d_in[13];
    const float* ln_g    = (const float*)d_in[14];
    const float* ln_b    = (const float*)d_in[15];
    const float* fW1     = (const float*)d_in[16];
    const float* fb1     = (const float*)d_in[17];
    const float* fW2     = (const float*)d_in[18];
    const float* fb2     = (const float*)d_in[19];
    const float* hW1     = (const float*)d_in[20];
    const float* hb1     = (const float*)d_in[21];
    const float* hW2     = (const float*)d_in[22];
    const float* hb2     = (const float*)d_in[23];
    float* out = (float*)d_out;

    char* ws = (char*)d_ws;
    // workspace layout (bytes)
    const size_t OFF_PACKED  = 0;                                 // 2*8*16*512 bf16 = 256 KB
    const size_t OFF_TARGETS = OFF_PACKED  + 262144;              // B*D f32 = 4 MB
    const size_t OFF_Q2      = OFF_TARGETS + (size_t)BB * DD * 4; // 4 MB
    const size_t OFF_WEFFP   = OFF_Q2      + (size_t)BB * DD * 4; // 8 KB
    const size_t OFF_WEFF2   = OFF_WEFFP   + HH * DD * 4;         // B*8*256 f32 = 32 MB
    const size_t OFF_E       = OFF_WEFF2   + (size_t)BB * HH * DD * 4; // T*16 f32 = 16 MB
    const size_t OFF_SEGMAX  = OFF_E       + (size_t)TT * 16 * 4; // B*16 u32
    const size_t OFF_DEN     = OFF_SEGMAX  + (size_t)BB * 16 * 4; // B*16 f32
    const size_t OFF_CTX     = OFF_DEN     + (size_t)BB * 16 * 4; // B*2*256 f32 = 8 MB

    unsigned short* packed  = (unsigned short*)(ws + OFF_PACKED);
    float*          targets = (float*)(ws + OFF_TARGETS);
    float*          q2      = (float*)(ws + OFF_Q2);
    float*          weffp   = (float*)(ws + OFF_WEFFP);
    float*          weff2   = (float*)(ws + OFF_WEFF2);
    float*          ebuf    = (float*)(ws + OFF_E);
    unsigned int*   segmax  = (unsigned int*)(ws + OFF_SEGMAX);
    float*          den     = (float*)(ws + OFF_DEN);
    float*          ctx     = (float*)(ws + OFF_CTX);

    hipLaunchKernelGGL(k_init, dim3(BB * 2 * DD / 256), dim3(256), 0, stream, segmax, den, ctx);
    hipLaunchKernelGGL(k_pack, dim3(131072 / 256), dim3(256), 0, stream, pma_Wv, cr_Wv, packed);
    hipLaunchKernelGGL(k_targets_q2, dim3(BB), dim3(256), 0, stream, emb, tgt_idx, cr_Wq, targets, q2);
    hipLaunchKernelGGL(k_weffp, dim3(1), dim3(256), 0, stream, pma_seed, pma_Wq, pma_Wk, weffp);
    hipLaunchKernelGGL(k_weff2, dim3(BB), dim3(256), 0, stream, q2, cr_Wk, weff2);
    hipLaunchKernelGGL(k_logits, dim3(TT * 16 / 256), dim3(256), 0, stream,
                       emb, p_idx, p_batch, weffp, weff2, ebuf, segmax);
    hipLaunchKernelGGL(k_expsum, dim3(TT * 16 / 256), dim3(256), 0, stream,
                       p_batch, ebuf, segmax, den);
    hipLaunchKernelGGL(k_wmma_v, dim3(TT / 16 / 4), dim3(128), 0, stream,
                       emb, p_idx, p_batch, p_w, packed, ebuf, den, ctx);
    hipLaunchKernelGGL(k_final, dim3(BB / RB), dim3(256), 0, stream,
                       targets, ctx, pma_Wo, cr_Wo, ln_g, ln_b,
                       fW1, fb1, fW2, fb2, hW1, hb1, hW2, hb2, out);
}
